// AttentionPooling_28853590295047
// MI455X (gfx1250) — matrix operand order repaired and checked by reading, not compiled, not run on hardware
//
#include <hip/hip_runtime.h>
#include <hip/hip_bf16.h>

typedef __attribute__((ext_vector_type(16))) _Float16 v16h;
typedef __attribute__((ext_vector_type(8)))  float    v8f;

#define NUM_SEGMENTS 1024
#define HID 256
#define H2  128

// Branch-free transcendentals on the native trans pipe.
__device__ __forceinline__ float fast_exp(float x) {
    // exp(x) = exp2(x * log2(e)); callers guarantee x <= ~0 (no overflow path).
    return __builtin_amdgcn_exp2f(x * 1.4426950408889634f);
}
__device__ __forceinline__ float fast_tanh(float x) {
    // tanh(x) = 1 - 2/(e^(2x)+1); exp2 overflow -> inf -> rcp -> 0 -> +1,
    // underflow -> 0 -> -1. No branches, no exec-mask divergence.
    float e = __builtin_amdgcn_exp2f(x * 2.8853900817779268f); // 2*log2(e)
    return 1.0f - 2.0f * __builtin_amdgcn_rcpf(e + 1.0f);
}

// ---------------------------------------------------------------------------
// Kernel 1: s[i] = tanh(x[i,:] @ W1 + b1) @ W2 + b2   via v_wmma_f32_16x16x32_f16
//
// ks loop kept rolled (unroll 1) so only one K-step's loads are in flight;
// 8 persistent accumulators; waves_per_eu(4) caps VGPRs <= 256 (no MSB
// switching) and keeps multi-wave occupancy for latency hiding.
// ---------------------------------------------------------------------------
__global__ __launch_bounds__(256)
__attribute__((amdgpu_waves_per_eu(4)))
void scores_kernel(
    const float* __restrict__ x, const float* __restrict__ W1,
    const float* __restrict__ b1, const float* __restrict__ W2,
    const float* __restrict__ b2, float* __restrict__ s, int num_nodes)
{
    // W1 as f16, pre-swizzled into B-fragment order:
    // frag[nt][ks][lane][j]  (8*8*32*16 f16 = 64 KB); each lane reads a
    // contiguous 32B run and consecutive lanes are consecutive in LDS.
    __shared__ _Float16 w1frag[8 * 8 * 32 * 16];

    for (int idx = threadIdx.x; idx < 8 * 8 * 32 * 16; idx += blockDim.x) {
        int j    = idx & 15;
        int lane = (idx >> 4) & 31;
        int ks   = (idx >> 9) & 7;
        int nt   = idx >> 12;
        // B-matrix (32x16, KxN) lane layout: lanes 0-15 -> K=0..7 & 16..23,
        // lanes 16-31 -> K=8..15 & 24..31; N = lane & 15.
        int N = (nt << 4) + (lane & 15);
        int K = (ks << 5) + ((j & 8) << 1) + ((lane >> 4) << 3) + (j & 7);
        w1frag[idx] = (_Float16)W1[K * H2 + N];
    }
    __syncthreads();

    const int lane = threadIdx.x & 31;
    const int wave = threadIdx.x >> 5;
    const int numTiles = (num_nodes + 15) >> 4;
    const int stride = gridDim.x * 8;
    const int r  = lane & 15;          // row within tile (A) / col (B,C)
    const int kb = (lane >> 4) << 3;   // A half-lane K base (0 or 8)
    const float b2v = b2[0];

#pragma unroll 1
    for (int tile = blockIdx.x * 8 + wave; tile < numTiles; tile += stride) {
        const int row0 = tile << 4;
        int arow = row0 + r;
        if (arow >= num_nodes) arow = num_nodes - 1;   // clamp; writes guarded
        const float* xrow = x + (size_t)arow * HID;

        v8f c[8];
#pragma unroll
        for (int nt = 0; nt < 8; ++nt) c[nt] = (v8f){};

#pragma unroll 1
        for (int ks = 0; ks < 8; ++ks) {
            // A fragment (16-bit A 16x32 layout): lanes 0-15: K=0..7 & 16..23;
            // lanes 16-31: K=8..15 & 24..31 (relative to this K-step).
            const float* pa = xrow + (ks << 5) + kb;
            v16h a;
#pragma unroll
            for (int j = 0; j < 8; ++j) a[j]     = (_Float16)pa[j];
#pragma unroll
            for (int j = 0; j < 8; ++j) a[8 + j] = (_Float16)pa[16 + j];

            const _Float16* pbase = &w1frag[(ks << 9) + (lane << 4)];
#pragma unroll
            for (int nt = 0; nt < 8; ++nt) {
                const _Float16* pb = pbase + (nt << 12);
                v16h b;
#pragma unroll
                for (int j = 0; j < 16; ++j) b[j] = pb[j];
                c[nt] = __builtin_amdgcn_wmma_f32_16x16x32_f16(
                            false, a, false, b, (short)0, c[nt], false, false);
            }
        }

        // Epilogue: h = tanh(c + b1[ncol]); rowAcc += h * W2[ncol]
        // C layout: VGPR v, lanes 0-15: M=v, N=lane; lanes 16-31: M=8+v, N=lane-16.
        float rowAcc[8];
#pragma unroll
        for (int v = 0; v < 8; ++v) rowAcc[v] = 0.0f;

#pragma unroll
        for (int nt = 0; nt < 8; ++nt) {
            const int ncol = (nt << 4) + r;
            const float b1v = b1[ncol];
            const float w2v = W2[ncol];
#pragma unroll
            for (int v = 0; v < 8; ++v)
                rowAcc[v] += fast_tanh(c[nt][v] + b1v) * w2v;
        }

        // Reduce over the 16 columns held across each 16-lane half.
#pragma unroll
        for (int v = 0; v < 8; ++v) {
            float acc = rowAcc[v];
#pragma unroll
            for (int mask = 8; mask >= 1; mask >>= 1)
                acc += __shfl_xor(acc, mask, 32);
            rowAcc[v] = acc;
        }
        const int mrow = row0 + ((lane >> 4) << 3); // lane0 -> rows v, lane16 -> rows 8+v
        if (r == 0) {
#pragma unroll
            for (int v = 0; v < 8; ++v) {
                int orow = mrow + v;
                if (orow < num_nodes) s[orow] = rowAcc[v] + b2v;
            }
        }
    }
}

// ---------------------------------------------------------------------------
// Kernel 2: segment offsets via binary search on the sorted batch array
// ---------------------------------------------------------------------------
__global__ void seg_bounds_kernel(const int* __restrict__ batch, int n,
                                  int* __restrict__ seg_start)
{
    int j = blockIdx.x * blockDim.x + threadIdx.x;
    if (j > NUM_SEGMENTS) return;
    if (j == NUM_SEGMENTS) { seg_start[j] = n; return; }
    int lo = 0, hi = n;
    while (lo < hi) {
        int mid = (lo + hi) >> 1;
        if (batch[mid] < j) lo = mid + 1; else hi = mid;
    }
    seg_start[j] = lo;
}

// ---------------------------------------------------------------------------
// Kernel 3: per-segment softmax + weighted column reduction (one block/segment)
// ---------------------------------------------------------------------------
__global__ __launch_bounds__(256) void pool_kernel(
    const float* __restrict__ x, const float* __restrict__ s,
    const int* __restrict__ seg_start, float* __restrict__ out)
{
    __shared__ float red[8];
    __shared__ float wbuf[256];
    __shared__ float bcast;

    const int seg = blockIdx.x;
    const int start = seg_start[seg];
    const int end   = seg_start[seg + 1];
    const int t = threadIdx.x;

    if (start >= end) { out[(size_t)seg * HID + t] = 0.0f; return; }

    // --- segment max ---
    float m = -3.402823466e38f;
    for (int i = start + t; i < end; i += 256) m = fmaxf(m, s[i]);
#pragma unroll
    for (int mask = 16; mask >= 1; mask >>= 1) m = fmaxf(m, __shfl_xor(m, mask, 32));
    if ((t & 31) == 0) red[t >> 5] = m;
    __syncthreads();
    if (t == 0) {
        float v = red[0];
        for (int i = 1; i < 8; ++i) v = fmaxf(v, red[i]);
        bcast = v;
    }
    __syncthreads();
    const float smax = bcast;

    // --- segment sum of exp (s - smax <= 0: fast_exp safe) ---
    float sum = 0.0f;
    for (int i = start + t; i < end; i += 256) sum += fast_exp(s[i] - smax);
#pragma unroll
    for (int mask = 16; mask >= 1; mask >>= 1) sum += __shfl_xor(sum, mask, 32);
    __syncthreads();
    if ((t & 31) == 0) red[t >> 5] = sum;
    __syncthreads();
    if (t == 0) {
        float v = 0.0f;
        for (int i = 0; i < 8; ++i) v += red[i];
        bcast = v;
    }
    __syncthreads();
    const float inv = 1.0f / (bcast + 1e-8f);

    // --- weighted sum; thread t owns column t (fully coalesced row reads) ---
    float acc = 0.0f;
    for (int base = start; base < end; base += 256) {
        int i = base + t;
        if (i < end) wbuf[t] = fast_exp(s[i] - smax) * inv;
        __syncthreads();
        const int cnt = (end - base < 256) ? (end - base) : 256;
        const float* xp = x + (size_t)base * HID + t;
        for (int rr = 0; rr < cnt; ++rr) {
            acc += xp[(size_t)rr * HID] * wbuf[rr];
            if ((rr & 7) == 0)
                __builtin_prefetch(xp + (size_t)(rr + 32) * HID, 0, 1);
        }
        __syncthreads();
    }
    out[(size_t)seg * HID + t] = acc;
}

// ---------------------------------------------------------------------------
extern "C" void kernel_launch(void* const* d_in, const int* in_sizes, int n_in,
                              void* d_out, int out_size, void* d_ws, size_t ws_size,
                              hipStream_t stream)
{
    const float* x     = (const float*)d_in[0];
    const int*   batch = (const int*)  d_in[1];
    const float* W1    = (const float*)d_in[2];
    const float* b1    = (const float*)d_in[3];
    const float* W2    = (const float*)d_in[4];
    const float* b2    = (const float*)d_in[5];
    float* out = (float*)d_out;

    const int num_nodes = in_sizes[0] / HID;

    // Workspace: scores (num_nodes f32) then segment offsets (1025 i32).
    float* s = (float*)d_ws;
    size_t s_bytes = ((size_t)num_nodes * sizeof(float) + 255) & ~(size_t)255;
    int* seg_off = (int*)((char*)d_ws + s_bytes);

    int numTiles = (num_nodes + 15) >> 4;
    int blocks = (numTiles + 7) >> 3;
    if (blocks > 2048) blocks = 2048;
    if (blocks < 1) blocks = 1;

    scores_kernel<<<blocks, 256, 0, stream>>>(x, W1, b1, W2, b2, s, num_nodes);
    seg_bounds_kernel<<<(NUM_SEGMENTS + 256) / 256, 256, 0, stream>>>(batch, num_nodes, seg_off);
    pool_kernel<<<NUM_SEGMENTS, 256, 0, stream>>>(x, s, seg_off, out);
}
